// CoAttentionNetwork_79233556676627
// MI455X (gfx1250) — matrix operand order repaired and checked by reading, not compiled, not run on hardware
//
#include <hip/hip_runtime.h>
#include <math.h>

// Problem constants (must match reference)
#define B_   64
#define N_   1024
#define T_   512
#define D_   384
#define K_   2
#define OUT_ 6

typedef __attribute__((ext_vector_type(16))) __bf16        v16bf;
typedef __attribute__((ext_vector_type(8)))  float         v8f;
typedef __attribute__((ext_vector_type(4)))  unsigned int  v4u;
typedef __attribute__((ext_vector_type(4)))  float         v4f;
typedef __attribute__((ext_vector_type(8)))  int           v8i;
typedef __attribute__((ext_vector_type(4)))  int           v4i;

#if __has_builtin(__builtin_amdgcn_tensor_load_to_lds) && \
    __has_builtin(__builtin_amdgcn_s_wait_tensorcnt)
#define USE_TDM 1
#endif

union BFrag { v4u q[2]; v16bf v; };

__device__ __forceinline__ unsigned short f2bf(float f) {
  unsigned int u = __float_as_uint(f);
  u += 0x7FFFu + ((u >> 16) & 1u);   // round-to-nearest-even
  return (unsigned short)(u >> 16);
}

// pack two f32 -> packed bf16x2 (lo | hi<<16), RNE
__device__ __forceinline__ unsigned int f2bf2(float lo, float hi) {
  unsigned int ul = __float_as_uint(lo);
  ul = (ul + 0x7FFFu + ((ul >> 16) & 1u)) >> 16;
  unsigned int uh = __float_as_uint(hi);
  uh = (uh + 0x7FFFu + ((uh >> 16) & 1u)) & 0xFFFF0000u;
  return ul | uh;
}

// ---------------------------------------------------------------------------
// Kernel 1: per-row rank-2 projections.
//   WsS[b,k,n] = Ws[k,:] . S[b,n,:]      (f32)
//   WcC[b,k,t] = Wc[k,:] . C[b,t,:]      (f32)
//   HcAcc[b,k,t] initialized to WcC (base term of Hc before atomics).
// ---------------------------------------------------------------------------
__global__ __launch_bounds__(256)
void prep_kernel(const float* __restrict__ S, const float* __restrict__ C,
                 const float* __restrict__ Ws, const float* __restrict__ Wc,
                 float* __restrict__ WsS, float* __restrict__ WcC,
                 float* __restrict__ HcAcc) {
  int wid  = (blockIdx.x * blockDim.x + threadIdx.x) >> 5;
  int lane = threadIdx.x & 31;
  if (wid < B_ * N_) {
    const float* row = S + (size_t)wid * D_;
    float a0 = 0.f, a1 = 0.f;
    for (int d = lane; d < D_; d += 32) {
      float s = row[d];
      a0 += Ws[d] * s;
      a1 += Ws[D_ + d] * s;
    }
    for (int off = 16; off > 0; off >>= 1) {
      a0 += __shfl_down(a0, off, 32);
      a1 += __shfl_down(a1, off, 32);
    }
    if (lane == 0) {
      int b = wid / N_, n = wid % N_;
      WsS[((size_t)b * K_ + 0) * N_ + n] = a0;
      WsS[((size_t)b * K_ + 1) * N_ + n] = a1;
    }
  } else if (wid < B_ * N_ + B_ * T_) {
    int w2 = wid - B_ * N_;
    const float* row = C + (size_t)w2 * D_;
    float a0 = 0.f, a1 = 0.f;
    for (int d = lane; d < D_; d += 32) {
      float c = row[d];
      a0 += Wc[d] * c;
      a1 += Wc[D_ + d] * c;
    }
    for (int off = 16; off > 0; off >>= 1) {
      a0 += __shfl_down(a0, off, 32);
      a1 += __shfl_down(a1, off, 32);
    }
    if (lane == 0) {
      int b = w2 / T_, t = w2 % T_;
      WcC[((size_t)b * K_ + 0) * T_ + t]   = a0;
      WcC[((size_t)b * K_ + 1) * T_ + t]   = a1;
      HcAcc[((size_t)b * K_ + 0) * T_ + t] = a0;
      HcAcc[((size_t)b * K_ + 1) * T_ + t] = a1;
    }
  }
}

// ---------------------------------------------------------------------------
// Kernel 2: CWl = C @ Wl as one flat GEMM  [B*T, D] x [D, D] -> bf16 out.
// Block tile 64(M) x 128(N), 8 waves (4x2), wave tile 16x64 (4 WMMA n-tiles).
// ---------------------------------------------------------------------------
__global__ __launch_bounds__(256)
void cwl_gemm(const float* __restrict__ Cg, const float* __restrict__ Wl,
              unsigned short* __restrict__ CWlh) {
  __shared__ __align__(16) unsigned short At[64][40];    // padded to 80B rows
  __shared__ __align__(16) unsigned short Bt[128][40];
  int tid  = threadIdx.x;
  int m0   = blockIdx.x * 64;
  int n0   = blockIdx.y * 128;
  int wave = tid >> 5, lane = tid & 31;
  int wm0  = (wave >> 1) * 16;
  int wn0  = (wave & 1) * 64;
  int r = lane & 15, hi = lane >> 4;

  v8f acc[4] = {};

  for (int kk = 0; kk < D_; kk += 32) {
    __syncthreads();
    { // stage A: 64x32 f32 -> bf16 packed u32 stores
      int row = tid >> 2, part = tid & 3;
      const v4f* src = (const v4f*)(Cg + (size_t)(m0 + row) * D_ + kk + part * 8);
      v4f x0 = src[0], x1 = src[1];
      unsigned int* dst = (unsigned int*)&At[row][part * 8];
      dst[0] = f2bf2(x0.x, x0.y);
      dst[1] = f2bf2(x0.z, x0.w);
      dst[2] = f2bf2(x1.x, x1.y);
      dst[3] = f2bf2(x1.z, x1.w);
    }
    // stage B transposed: Bt[n][k] = Wl[(kk+k)*D + n0+n]
    for (int i = 0; i < 16; i++) {
      int idx = tid + i * 256;
      int n = idx & 127, k = idx >> 7;
      Bt[n][k] = f2bf(Wl[(size_t)(kk + k) * D_ + n0 + n]);
    }
    __syncthreads();
    BFrag fa;
    fa.q[0] = *(const v4u*)&At[wm0 + r][hi * 8];
    fa.q[1] = *(const v4u*)&At[wm0 + r][hi * 8 + 16];
#pragma unroll
    for (int j = 0; j < 4; j++) {
      BFrag fb;
      const unsigned short* bp = &Bt[wn0 + j * 16 + r][hi * 8];
      fb.q[0] = *(const v4u*)bp;
      fb.q[1] = *(const v4u*)(bp + 16);
      acc[j] = __builtin_amdgcn_wmma_f32_16x16x32_bf16(
          false, fa.v, false, fb.v, (short)0, acc[j], false, false);
    }
  }
#pragma unroll
  for (int j = 0; j < 4; j++)
#pragma unroll
    for (int rr = 0; rr < 8; rr++) {
      int m = m0 + wm0 + rr + hi * 8;
      int n = n0 + wn0 + j * 16 + r;
      CWlh[(size_t)m * D_ + n] = f2bf(acc[j][rr]);
    }
}

// ---------------------------------------------------------------------------
// Kernel 3 (fused core): per (batch b, 128-wide n-strip).
// F = tanh(CWl[t-chunk] @ S[strip]^T) in WMMA accumulators, folded directly
// into hs/hc rank-2 contractions; F never touches memory.
// A-tile (bf16 copy) is staged by the Tensor Data Mover when available.
// ---------------------------------------------------------------------------
__global__ __launch_bounds__(256)
void coattn_main(const unsigned short* __restrict__ CWlh,
                 const float* __restrict__ Sg,
                 const float* __restrict__ WsS, const float* __restrict__ WcC,
                 float* __restrict__ HsOut, float* __restrict__ HcAcc) {
  __shared__ __align__(16) unsigned short At[64][40];   // 80B rows: 64B data + 16B pad
  __shared__ __align__(16) unsigned short Bt[128][40];
  __shared__ float hsAcc[2][128];
  __shared__ float hcAcc[2][64];
  __shared__ float wcc_s[2][64];
  __shared__ float wss_s[2][128];

  int tid = threadIdx.x;
  int n0  = blockIdx.x * 128;
  int b   = blockIdx.y;
  int wave = tid >> 5, lane = tid & 31;
  int wm0  = (wave >> 1) * 16, wn0 = (wave & 1) * 64;
  int r = lane & 15, hi = lane >> 4;

  { // stage WsS for this strip; zero hsAcc
    int k = tid >> 7, nl = tid & 127;
    wss_s[k][nl] = WsS[((size_t)b * K_ + k) * N_ + n0 + nl];
    hsAcc[k][nl] = 0.f;
  }

  for (int tc = 0; tc < T_; tc += 64) {
    if (tid < 128) {
      int k = tid >> 6, tl = tid & 63;
      wcc_s[k][tl] = WcC[((size_t)b * K_ + k) * T_ + tc + tl];
      hcAcc[k][tl] = 0.f;
    }
    v8f acc[4] = {};
    for (int kk = 0; kk < D_; kk += 32) {
      __syncthreads();
#ifdef USE_TDM
      // ---- TDM: DMA the 64x32 bf16 A-tile of CWl into LDS (wave 0 only).
      // D# tile: dim0=32 elems (64B rows), dim1=64 rows, stride=D elems;
      // LDS padding 4 DWORDs every 16 DWORDs matches the 80B padded rows.
      if (wave == 0) {
        unsigned long long ga = (unsigned long long)(const void*)CWlh +
                                2ull * ((size_t)(b * T_ + tc) * D_ + kk);
        unsigned int lds = (unsigned int)(size_t)&At[0][0];
        v4u g0;
        g0.x = 1u;                                   // count=1 (valid user D#)
        g0.y = lds;                                  // lds_addr
        g0.z = (unsigned int)(ga & 0xFFFFFFFFu);     // global_addr[31:0]
        g0.w = (unsigned int)((ga >> 32) & 0x01FFFFFFu) | (2u << 30); // +type=2
        v8i g1;
        g1[0] = (int)0x06D10000;        // data_size=2B, pad_en, ivl=16dw, amt=4dw
        g1[1] = (int)(((unsigned)D_ & 0xFFFFu) << 16);  // tensor_dim0 = 384 (lo16)
        g1[2] = (int)0x80000000u;       // tensor_dim1 = 32768 (lo16 @ [95:80])
        g1[3] = (int)(32u << 16);       // tile_dim0 = 32 @ [127:112]
        g1[4] = 64;                     // tile_dim1 = 64 @ [143:128]
        g1[5] = D_;                     // tensor_dim0_stride = 384
        g1[6] = 0;
        g1[7] = 0;
        v4i gz4 = {0, 0, 0, 0};
        v8i gz8 = {0, 0, 0, 0, 0, 0, 0, 0};
        __builtin_amdgcn_tensor_load_to_lds(g0, g1, gz4, gz4, gz8, 0);
      }
#else
      { // manual A staging: 64x32 bf16 tile, 16B copies
        int row = tid >> 2, part = tid & 3;
        const v4u* src =
            (const v4u*)(CWlh + ((size_t)(b * T_ + tc + row) * D_ + kk + part * 8));
        *(v4u*)&At[row][part * 8] = *src;
      }
#endif
      { // stage B: Bt[n][k] = bf16(S[b, n0+n, kk+k]); packed u32 stores
        int n = tid >> 1, kb = (tid & 1) * 16;
        const float* sp = Sg + ((size_t)(b * N_ + n0 + n) * D_ + kk + kb);
        const v4f* src = (const v4f*)sp;
        v4f x0 = src[0], x1 = src[1], x2 = src[2], x3 = src[3];
        unsigned int* dst = (unsigned int*)&Bt[n][kb];
        dst[0] = f2bf2(x0.x, x0.y); dst[1] = f2bf2(x0.z, x0.w);
        dst[2] = f2bf2(x1.x, x1.y); dst[3] = f2bf2(x1.z, x1.w);
        dst[4] = f2bf2(x2.x, x2.y); dst[5] = f2bf2(x2.z, x2.w);
        dst[6] = f2bf2(x3.x, x3.y); dst[7] = f2bf2(x3.z, x3.w);
        if (kk + 32 < D_) __builtin_prefetch(sp + 32, 0, 0);   // next k-slice
      }
#ifdef USE_TDM
      if (wave == 0) __builtin_amdgcn_s_wait_tensorcnt(0);
#endif
      __syncthreads();
      BFrag fa;
      fa.q[0] = *(const v4u*)&At[wm0 + r][hi * 8];
      fa.q[1] = *(const v4u*)&At[wm0 + r][hi * 8 + 16];
#pragma unroll
      for (int j = 0; j < 4; j++) {
        BFrag fb;
        const unsigned short* bp = &Bt[wn0 + j * 16 + r][hi * 8];
        fb.q[0] = *(const v4u*)bp;
        fb.q[1] = *(const v4u*)(bp + 16);
        acc[j] = __builtin_amdgcn_wmma_f32_16x16x32_bf16(
            false, fa.v, false, fb.v, (short)0, acc[j], false, false);
      }
    }
    // Epilogue: tanh in registers, fold F into hs/hc accumulators.
#pragma unroll
    for (int j = 0; j < 4; j++) {
      float f[8];
      int nl = wn0 + j * 16 + r;
#pragma unroll
      for (int rr = 0; rr < 8; rr++) f[rr] = tanhf(acc[j][rr]);
#pragma unroll
      for (int k = 0; k < 2; k++) {   // hs: reduce over this wave's 8 t-rows
        float s = 0.f;
#pragma unroll
        for (int rr = 0; rr < 8; rr++) s += wcc_s[k][wm0 + hi * 8 + rr] * f[rr];
        atomicAdd(&hsAcc[k][nl], s);
      }
#pragma unroll
      for (int k = 0; k < 2; k++) {   // hc: butterfly-reduce over 16 columns
        float w = wss_s[k][nl];
#pragma unroll
        for (int rr = 0; rr < 8; rr++) {
          float v = f[rr] * w;
          v += __shfl_xor(v, 1, 32);
          v += __shfl_xor(v, 2, 32);
          v += __shfl_xor(v, 4, 32);
          v += __shfl_xor(v, 8, 32);
          if ((lane & 15) == 0)
            atomicAdd(&hcAcc[k][wm0 + hi * 8 + rr], v);
        }
      }
    }
    __syncthreads();
    if (tid < 128) {  // flush this t-chunk's Hc partials (cross-strip sum)
      int k = tid >> 6, tl = tid & 63;
      atomicAdd(&HcAcc[((size_t)b * K_ + k) * T_ + tc + tl], hcAcc[k][tl]);
    }
    __syncthreads();
  }
  { // Hs finished locally: base term + tanh, plain store
    int k = tid >> 7, nl = tid & 127;
    HsOut[((size_t)b * K_ + k) * N_ + n0 + nl] =
        tanhf(wss_s[k][nl] + hsAcc[k][nl]);
  }
}

// ---------------------------------------------------------------------------
// Kernel 4: per-batch softmaxes + attention-weighted sums + final linear.
// ---------------------------------------------------------------------------
__device__ __forceinline__ void softmax_shared(float* v, int n, float* red, int tid) {
  float m = -1e30f;
  for (int i = tid; i < n; i += 256) m = fmaxf(m, v[i]);
  red[tid] = m; __syncthreads();
  for (int s = 128; s > 0; s >>= 1) {
    if (tid < s) red[tid] = fmaxf(red[tid], red[tid + s]);
    __syncthreads();
  }
  float mx = red[0]; __syncthreads();
  float sum = 0.f;
  for (int i = tid; i < n; i += 256) { float e = __expf(v[i] - mx); v[i] = e; sum += e; }
  red[tid] = sum; __syncthreads();
  for (int s = 128; s > 0; s >>= 1) {
    if (tid < s) red[tid] += red[tid + s];
    __syncthreads();
  }
  float inv = 1.f / red[0]; __syncthreads();
  for (int i = tid; i < n; i += 256) v[i] *= inv;
  __syncthreads();
}

__global__ __launch_bounds__(256)
void finalize_kernel(const float* __restrict__ Sg, const float* __restrict__ Cg,
                     const float* __restrict__ HsOut, const float* __restrict__ HcAcc,
                     const float* __restrict__ Whs, const float* __restrict__ Whc,
                     const float* __restrict__ fc_w, const float* __restrict__ fc_b,
                     float* __restrict__ out) {
  __shared__ float ls[N_];
  __shared__ float lc[T_];
  __shared__ float red[256];
  __shared__ float co[2 * D_];
  int tid = threadIdx.x;
  int b = blockIdx.x;
  float whs0 = Whs[0], whs1 = Whs[1];
  float whc0 = Whc[0], whc1 = Whc[1];
  for (int n = tid; n < N_; n += 256)
    ls[n] = whs0 * HsOut[((size_t)b * K_ + 0) * N_ + n] +
            whs1 * HsOut[((size_t)b * K_ + 1) * N_ + n];
  for (int t = tid; t < T_; t += 256)
    lc[t] = whc0 * tanhf(HcAcc[((size_t)b * K_ + 0) * T_ + t]) +
            whc1 * tanhf(HcAcc[((size_t)b * K_ + 1) * T_ + t]);
  __syncthreads();
  softmax_shared(ls, N_, red, tid);   // -> As
  softmax_shared(lc, T_, red, tid);   // -> Ac
  for (int d = tid; d < D_; d += 256) {   // co_s[d]
    float a = 0.f;
    for (int n = 0; n < N_; n++) a += ls[n] * Sg[((size_t)b * N_ + n) * D_ + d];
    co[d] = a;
  }
  for (int d = tid; d < D_; d += 256) {   // co_c[d]
    float a = 0.f;
    for (int t = 0; t < T_; t++) a += lc[t] * Cg[((size_t)b * T_ + t) * D_ + d];
    co[D_ + d] = a;
  }
  __syncthreads();
  if (tid < OUT_) {
    float a = fc_b[tid];
    for (int i = 0; i < 2 * D_; i++) a += fc_w[(size_t)tid * (2 * D_) + i] * co[i];
    out[b * OUT_ + tid] = a;
  }
}

// ---------------------------------------------------------------------------
// Workspace layout (bytes), total ~26.7 MB:
//   [0)            CWlh  : B*T*D bf16 = 25,165,824
//   [25165824)     WsS   : B*K*N f32  =    524,288
//   [25690112)     WcC   : B*K*T f32  =    262,144
//   [25952256)     HcAcc : B*K*T f32  =    262,144
//   [26214400)     HsOut : B*K*N f32  =    524,288
// ---------------------------------------------------------------------------
extern "C" void kernel_launch(void* const* d_in, const int* in_sizes, int n_in,
                              void* d_out, int out_size, void* d_ws, size_t ws_size,
                              hipStream_t stream) {
  const float* S    = (const float*)d_in[0];
  const float* C    = (const float*)d_in[1];
  const float* Wl   = (const float*)d_in[2];
  const float* Ws   = (const float*)d_in[3];
  const float* Wc   = (const float*)d_in[4];
  const float* Whs  = (const float*)d_in[5];
  const float* Whc  = (const float*)d_in[6];
  const float* fc_w = (const float*)d_in[7];
  const float* fc_b = (const float*)d_in[8];
  float* out = (float*)d_out;

  char* ws = (char*)d_ws;
  unsigned short* CWlh = (unsigned short*)(ws + 0);
  float* WsS   = (float*)(ws + 25165824);
  float* WcC   = (float*)(ws + 25690112);
  float* HcAcc = (float*)(ws + 25952256);
  float* HsOut = (float*)(ws + 26214400);

  prep_kernel<<<dim3((B_ * N_ + B_ * T_) / 8), 256, 0, stream>>>(
      S, C, Ws, Wc, WsS, WcC, HcAcc);
  cwl_gemm<<<dim3((B_ * T_) / 64, D_ / 128), 256, 0, stream>>>(C, Wl, CWlh);
  coattn_main<<<dim3(N_ / 128, B_), 256, 0, stream>>>(
      CWlh, S, WsS, WcC, HsOut, HcAcc);
  finalize_kernel<<<dim3(B_), 256, 0, stream>>>(
      S, C, HsOut, HcAcc, Whs, Whc, fc_w, fc_b, out);
}
